// Quantizer_7687991460418
// MI455X (gfx1250) — matrix operand order repaired and checked by reading, not compiled, not run on hardware
//
#include <hip/hip_runtime.h>

typedef _Float16 v16h __attribute__((ext_vector_type(16)));
typedef _Float16 v8h  __attribute__((ext_vector_type(8)));
typedef float    v8f  __attribute__((ext_vector_type(8)));

#define N_ROWS   131072          // 4*32*32*32 flattened spatial positions
#define CDIM     64              // embed dim
#define NCODE    512             // codebook entries
#define SPAT     32768           // 32*32*32 (per-batch spatial)
#define ROWS_PER_BLOCK 128
#define NBLOCKS  (N_ROWS / ROWS_PER_BLOCK)   // 1024
#define CHUNK    256             // codebook entries staged in LDS at a time
#define WSTRIDE  72              // padded f16 row stride (144 B, 16B-aligned, bank-friendly)
#define Q_ELEMS  (N_ROWS * CDIM) // 8388608
#define IDX_OFF  Q_ELEMS
#define LOSS_OFF (Q_ELEMS + N_ROWS)

__device__ __forceinline__ void load_bfrag(const _Float16* wrow, int k0b,
                                           v16h& b0, v16h& b1) {
    const v8h* p0 = (const v8h*)(wrow + k0b);        // K = k0b..k0b+15
    const v8h* p1 = (const v8h*)(wrow + 32 + k0b);   // K = 32+k0b..
    v8h q0 = p0[0], q1 = p0[1], q2 = p1[0], q3 = p1[1];
#pragma unroll
    for (int v = 0; v < 8; ++v) {
        b0[v] = q0[v]; b0[v + 8] = q1[v];
        b1[v] = q2[v]; b1[v + 8] = q3[v];
    }
}

__global__ __launch_bounds__(256) void vq_main(const float* __restrict__ x,
                                               const float* __restrict__ weight,
                                               float* __restrict__ out,
                                               float* __restrict__ partial) {
    __shared__ _Float16 wh[CHUNK * WSTRIDE];     // 36 KB fp16 codebook chunk
    __shared__ float    hnorm[CHUNK];            // 1 KB  0.5*||w||^2 per entry
    __shared__ int      idx_lds[ROWS_PER_BLOCK]; // argmin per row
    __shared__ float    red[256];                // loss reduction

    const int t    = threadIdx.x;
    const int lane = t & 31;
    const int wave = t >> 5;
    const int n0   = blockIdx.x * ROWS_PER_BLOCK;

    const int ncol = lane & 15;        // column within 16-wide N tile
    const int hiH  = lane >> 4;        // 0: lanes 0-15, 1: lanes 16-31

    // ---- A fragments (16x32 f16, K=0..31 and K=32..63) for this wave's 16-row tile ----
    // x layout is [b, c, 32768]: for fixed c, consecutive rows are contiguous -> coalesced.
    const int row = (wave << 4) + ncol;            // 0..127 within block
    const int n   = n0 + row;
    const int b   = n >> 15;
    const int sp  = n & (SPAT - 1);
    const float* px = x + (size_t)b * (CDIM * SPAT) + sp;

    v16h a0, a1;
#pragma unroll
    for (int v = 0; v < 16; ++v) {
        // A k order used here (contiguous, same as B): lanes 0-15 hold K 0-15, lanes 16-31 hold K 16-31
        int k = v + (hiH ? 16 : 0);
        a0[v] = (_Float16)px[(size_t)k * SPAT];
        a1[v] = (_Float16)px[(size_t)(k + 32) * SPAT];
    }

    float bestv[8];
    int   besti[8];
#pragma unroll
    for (int r = 0; r < 8; ++r) { bestv[r] = -3.402823e38f; besti[r] = 0; }

    const int k0b = hiH ? 16 : 0;      // B-matrix: lanes 0-15 K=0-15, lanes 16-31 K=16-31

    for (int ch = 0; ch < NCODE / CHUNK; ++ch) {
        // ---- stage codebook chunk f32 -> f16 into LDS + half-norms ----
        for (int i = t; i < CHUNK * CDIM; i += 256) {
            int e = i >> 6, c = i & 63;
            wh[e * WSTRIDE + c] = (_Float16)weight[(ch * CHUNK + e) * CDIM + c];
        }
        {
            int e = t;  // 256 threads, 256 entries
            const float* wr = weight + (ch * CHUNK + e) * CDIM;
            float s = 0.f;
#pragma unroll
            for (int c = 0; c < CDIM; ++c) s += wr[c] * wr[c];
            hnorm[e] = 0.5f * s;
        }
        __syncthreads();

        // two independent accumulator chains per iteration -> WMMA latency hidden
        for (int jt2 = 0; jt2 < CHUNK / 32; ++jt2) {
            const int jlA = (jt2 << 5) + ncol;         // entry (tile A) within chunk
            const int jlB = jlA + 16;                  // entry (tile B) within chunk
            const int jgA = ch * CHUNK + jlA;
            const int jgB = jgA + 16;

            v16h bA0, bA1, bB0, bB1;
            load_bfrag(&wh[jlA * WSTRIDE], k0b, bA0, bA1);
            load_bfrag(&wh[jlB * WSTRIDE], k0b, bB0, bB1);

            // fold -0.5*||w_j||^2 into the accumulators (same column j for all 8 slots)
            float hnA = hnorm[jlA];
            float hnB = hnorm[jlB];
            v8f accA, accB;
#pragma unroll
            for (int r = 0; r < 8; ++r) { accA[r] = -hnA; accB[r] = -hnB; }

            accA = __builtin_amdgcn_wmma_f32_16x16x32_f16(false, a0, false, bA0,
                                                          (short)0, accA, false, false);
            accB = __builtin_amdgcn_wmma_f32_16x16x32_f16(false, a0, false, bB0,
                                                          (short)0, accB, false, false);
            accA = __builtin_amdgcn_wmma_f32_16x16x32_f16(false, a1, false, bA1,
                                                          (short)0, accA, false, false);
            accB = __builtin_amdgcn_wmma_f32_16x16x32_f16(false, a1, false, bB1,
                                                          (short)0, accB, false, false);
#pragma unroll
            for (int r = 0; r < 8; ++r) {
                if (accA[r] > bestv[r]) { bestv[r] = accA[r]; besti[r] = jgA; }
            }
#pragma unroll
            for (int r = 0; r < 8; ++r) {
                if (accB[r] > bestv[r]) { bestv[r] = accB[r]; besti[r] = jgB; }
            }
        }
        __syncthreads();   // before next chunk overwrites wh
    }

    // ---- argmax butterfly across the 16 column-lanes (ties -> lowest index) ----
#pragma unroll
    for (int r = 0; r < 8; ++r) {
#pragma unroll
        for (int off = 1; off < 16; off <<= 1) {
            float ov = __shfl_xor(bestv[r], off, 32);
            int   oi = __shfl_xor(besti[r], off, 32);
            if (ov > bestv[r] || (ov == bestv[r] && oi < besti[r])) {
                bestv[r] = ov; besti[r] = oi;
            }
        }
    }
    if (ncol == 0) {
        // C layout: slot r, lanes 0-15 -> M=r, lanes 16-31 -> M=r+8
#pragma unroll
        for (int r = 0; r < 8; ++r)
            idx_lds[(wave << 4) + (hiH << 3) + r] = besti[r];
    }
    __syncthreads();

    // ---- phase 2: indices out, quantized out (exact f32 gather), fused loss ----
    if (t < ROWS_PER_BLOCK)
        out[IDX_OFF + n0 + t] = (float)idx_lds[t];

    float lsum = 0.f;
    {
        const int r2 = t & 127;
        const int chalf = t >> 7;                    // two channels per iteration
        const int n2 = n0 + r2;
        const int b2 = n2 >> 15;
        const int s2 = n2 & (SPAT - 1);
        const size_t base = (size_t)b2 * (CDIM * SPAT) + s2;
        const float* wr = weight + idx_lds[r2] * CDIM;
#pragma unroll 4
        for (int cc = 0; cc < 32; ++cc) {
            int c = (cc << 1) + chalf;
            float wq = wr[c];                        // L2-hot 128 KB gather
            size_t a = base + (size_t)c * SPAT;
            float xv = x[a];                         // streaming reload
            out[a] = wq;                             // coalesced (rows contiguous)
            float d = wq - xv;
            lsum += d * d;
        }
    }
    red[t] = lsum;
    __syncthreads();
    for (int st = 128; st > 0; st >>= 1) {
        if (t < st) red[t] += red[t + st];
        __syncthreads();
    }
    if (t == 0) partial[blockIdx.x] = red[0];
}

__global__ __launch_bounds__(256) void vq_finish(const float* __restrict__ partial,
                                                 float* __restrict__ out) {
    __shared__ float red[256];
    float s = 0.f;
    for (int i = threadIdx.x; i < NBLOCKS; i += 256) s += partial[i];
    red[threadIdx.x] = s;
    __syncthreads();
    for (int st = 128; st > 0; st >>= 1) {
        if (threadIdx.x < st) red[threadIdx.x] += red[threadIdx.x + st];
        __syncthreads();
    }
    if (threadIdx.x == 0)
        out[LOSS_OFF] = 0.25f * red[0] / (float)Q_ELEMS;
}

extern "C" void kernel_launch(void* const* d_in, const int* in_sizes, int n_in,
                              void* d_out, int out_size, void* d_ws, size_t ws_size,
                              hipStream_t stream) {
    const float* x      = (const float*)d_in[0];   // [4,64,32,32,32] f32
    const float* weight = (const float*)d_in[1];   // [512,64] f32
    float* out     = (float*)d_out;                // quantized | idx(as f32) | loss
    float* partial = (float*)d_ws;                 // NBLOCKS f32 partials

    vq_main<<<NBLOCKS, 256, 0, stream>>>(x, weight, out, partial);
    vq_finish<<<1, 256, 0, stream>>>(partial, out);
}
